// Attention_73813307949260
// MI455X (gfx1250) — matrix operand order repaired and checked by reading, not compiled, not run on hardware
//
#include <hip/hip_runtime.h>

// ---------------------------------------------------------------------------
// Sizes (fixed by the reference): B=8, N_EXAMPLES=10 -> BN=80, L=256, S=512,
// H=1024, 3H=3072.
// ---------------------------------------------------------------------------
constexpr int BN = 80;
constexpr int L  = 256;
constexpr int SSET = 512;
constexpr int H  = 1024;
constexpr int H3 = 3072;

typedef __attribute__((ext_vector_type(16))) __bf16 v16bf;
typedef __attribute__((ext_vector_type(8)))  float  v8f;

union Frag32 {
  v16bf v;
  uint4 u[2];
};
static_assert(sizeof(Frag32) == 32, "frag size");

__device__ __forceinline__ v8f vzero8() {
  v8f z = {0.f, 0.f, 0.f, 0.f, 0.f, 0.f, 0.f, 0.f};
  return z;
}

// ---------------------------------------------------------------------------
// fp32 -> bf16 elementwise convert
// ---------------------------------------------------------------------------
__global__ void k_cvt(const float* __restrict__ in, __bf16* __restrict__ out, long n) {
  long i = (long)blockIdx.x * blockDim.x + threadIdx.x;
  if (i < n) out[i] = (__bf16)in[i];
}

// fp32 [nb][s][h] -> bf16 [nb][h][s]  (V^T staging for the P@V matmuls)
__global__ void k_cvtT(const float* __restrict__ in, __bf16* __restrict__ out,
                       int s, int h, long n) {
  long i = (long)blockIdx.x * blockDim.x + threadIdx.x;
  if (i >= n) return;
  int hh = (int)(i % h);
  long t = i / h;
  int ss = (int)(t % s);
  int bb = (int)(t / s);
  out[((long)bb * h + hh) * (long)s + ss] = (__bf16)in[i];
}

// ---------------------------------------------------------------------------
// Fused attention branch:
//   scores = Q @ K^T  (k-dim H, bf16 WMMA, fp32 acc; Q tile staged in LDS)
//   probs  = softmax(scores)  -> fp32 to d_out, bf16 to LDS
//   ctx    = probs @ V        (k-dim SS, B-fragments from pre-transposed V^T)
// One workgroup = (batch b, 16-row tile of L). 4 waves, 128 threads.
// SS = 256 (example branch) or 512 (set branch).
// ---------------------------------------------------------------------------
template <int SS>
__global__ void k_attn(const __bf16* __restrict__ Qb,   // [BN][L][H]
                       const __bf16* __restrict__ Kb,   // [nbk][SS][H]
                       const __bf16* __restrict__ VT,   // [nbk][H][SS]
                       float*  __restrict__ attn_out,   // [BN][L][SS]
                       __bf16* __restrict__ c_out,      // [BN][L][H]
                       int kb_div) {
  constexpr int TPW = (SS / 16) / 4;  // score col-tiles per wave (4 or 8)

  const int wg   = blockIdx.x;
  const int b    = wg >> 4;       // L/16 == 16 row tiles
  const int l0   = (wg & 15) * 16;
  const int bk   = b / kb_div;
  const int tid  = threadIdx.x;
  const int wave = tid >> 5;
  const int lane = tid & 31;
  const int arow = lane & 15;     // A-row / B-column lane index
  const int agrp = lane >> 4;     // half-wave selector (k-group)

  extern __shared__ char smem[];
  __bf16* Qlds = (__bf16*)smem;                             // [16][H] 32 KB
  float*  sc   = (float*)(smem + 16 * H * 2);               // [16][SS]
  __bf16* P    = (__bf16*)(smem + 16 * H * 2 + 16 * SS * 4);// [16][SS]
  float*  red  = (float*)(smem + 16 * H * 2 + 16 * SS * 6); // [16][8]

  const __bf16* Kbase = Kb + (long)bk * SS * H;

  // ---- stage Q tile (16 x 1024 bf16 = 32 KB) in LDS once ----
  {
    uint4* q4d = (uint4*)Qlds;
    const uint4* q4s = (const uint4*)(Qb + ((long)b * L + l0) * H);
    for (int i = tid; i < 2048; i += 128) q4d[i] = q4s[i];  // rows contiguous
  }
  __syncthreads();

  // ---- phase 1: scores = Q @ K^T ----
  v8f acc[TPW];
#pragma unroll
  for (int t = 0; t < TPW; ++t) acc[t] = vzero8();

  for (int k0 = 0; k0 < H; k0 += 32) {
    Frag32 a;
    const __bf16* ap = Qlds + arow * H + k0 + agrp * 8;
    a.u[0] = *(const uint4*)ap;
    a.u[1] = *(const uint4*)(ap + 16);
#pragma unroll
    for (int t = 0; t < TPW; ++t) {
      const int s0 = (wave + 4 * t) * 16;
      Frag32 bf;
      const __bf16* bp = Kbase + (long)(s0 + arow) * H + k0 + agrp * 8;
      bf.u[0] = *(const uint4*)bp;
      bf.u[1] = *(const uint4*)(bp + 16);
      acc[t] = __builtin_amdgcn_wmma_f32_16x16x32_bf16(
          false, a.v, false, bf.v, (short)0, acc[t], false, false);
    }
  }

#pragma unroll
  for (int t = 0; t < TPW; ++t) {
    const int s0 = (wave + 4 * t) * 16;
#pragma unroll
    for (int r = 0; r < 8; ++r) {
      const int row = r + agrp * 8;             // C/D layout: M = r (+8 for hi lanes)
      sc[row * SS + s0 + arow] = acc[t][r];
    }
  }
  __syncthreads();

  // ---- phase 2: row softmax (8 threads per row) ----
  {
    const int row = tid >> 3;
    const int sub = tid & 7;
    float m = -3.4e38f;
    for (int s = sub; s < SS; s += 8) m = fmaxf(m, sc[row * SS + s]);
    red[row * 8 + sub] = m;
    __syncthreads();
    float rm = red[row * 8];
#pragma unroll
    for (int j = 1; j < 8; ++j) rm = fmaxf(rm, red[row * 8 + j]);
    float sum = 0.f;
    for (int s = sub; s < SS; s += 8) {
      float e = __expf(sc[row * SS + s] - rm);
      sc[row * SS + s] = e;
      sum += e;
    }
    __syncthreads();
    red[row * 8 + sub] = sum;
    __syncthreads();
    float rs = 0.f;
#pragma unroll
    for (int j = 0; j < 8; ++j) rs += red[row * 8 + j];
    const float inv = 1.f / rs;
    float* arow_out = attn_out + ((long)b * L + l0 + row) * SS;
    for (int s = sub; s < SS; s += 8) {
      const float p = sc[row * SS + s] * inv;
      arow_out[s]   = p;             // fp32 attn matrix (reference output 2/3)
      P[row * SS + s] = (__bf16)p;   // bf16 probs for the PV matmul
    }
  }
  __syncthreads();

  // ---- phase 3: ctx = P @ V  (B fragments from V^T, contiguous rows) ----
  const __bf16* VTb = VT + (long)bk * H * SS;
  for (int i = 0; i < 16; ++i) {                 // 64 H-tiles / 4 waves
    const int h0 = (wave + 4 * i) * 16;
    v8f c = vzero8();
#pragma unroll 4
    for (int k0 = 0; k0 < SS; k0 += 32) {
      Frag32 a;
      const __bf16* ap = P + arow * SS + k0 + agrp * 8;
      a.u[0] = *(const uint4*)ap;
      a.u[1] = *(const uint4*)(ap + 16);
      Frag32 bf;
      const __bf16* bp = VTb + (long)(h0 + arow) * SS + k0 + agrp * 8;
      bf.u[0] = *(const uint4*)bp;
      bf.u[1] = *(const uint4*)(bp + 16);
      c = __builtin_amdgcn_wmma_f32_16x16x32_bf16(
          false, a.v, false, bf.v, (short)0, c, false, false);
    }
    __bf16* crow = c_out + ((long)b * L + l0 + agrp * 8) * H + h0 + arow;
#pragma unroll
    for (int r = 0; r < 8; ++r) crow[(long)r * H] = (__bf16)c[r];
  }
}

// ---------------------------------------------------------------------------
// Final fused output GEMM, M-tile = 32 rows for 2x B-fragment reuse:
//   all_info = [output | ex_c | set_c] staged in LDS (32 x 3072 bf16 = 192 KB)
//   out = tanh(all_info @ W_out^T + b_out)   (k-dim 3072)
// One workgroup = (batch, 32-row tile); 4 waves sweep 64 H-tiles; each B
// fragment feeds two WMMAs (rows 0-15 and 16-31).
// ---------------------------------------------------------------------------
__global__ void k_out(const __bf16* __restrict__ Qb,
                      const __bf16* __restrict__ ExC,
                      const __bf16* __restrict__ SetC,
                      const __bf16* __restrict__ Wb,   // [H][3H] bf16
                      const float*  __restrict__ bias, // [H]
                      float* __restrict__ out) {       // [BN][L][H]
  const int wg   = blockIdx.x;
  const int b    = wg >> 3;       // L/32 == 8 row tiles
  const int l0   = (wg & 7) * 32;
  const int tid  = threadIdx.x;
  const int wave = tid >> 5;
  const int lane = tid & 31;
  const int arow = lane & 15;
  const int agrp = lane >> 4;

  extern __shared__ char smem[];
  __bf16* A  = (__bf16*)smem;   // [32][3072]
  uint4*  A4 = (uint4*)smem;

  const long rowbase = ((long)b * L + l0) * H;
  const uint4* q4 = (const uint4*)(Qb + rowbase);
  const uint4* e4 = (const uint4*)(ExC + rowbase);
  const uint4* s4 = (const uint4*)(SetC + rowbase);
  for (int i = tid; i < 4096; i += 128) {   // 32 rows * 128 uint4 per source
    const int r = i >> 7;
    const int c = i & 127;
    A4[r * 384 + c]       = q4[r * 128 + c];
    A4[r * 384 + 128 + c] = e4[r * 128 + c];
    A4[r * 384 + 256 + c] = s4[r * 128 + c];
  }
  __syncthreads();

  for (int i = 0; i < 16; ++i) {
    const int h0 = (wave + 4 * i) * 16;
    v8f c0 = vzero8();
    v8f c1 = vzero8();
    for (int k0 = 0; k0 < H3; k0 += 32) {
      Frag32 bf;                     // one B fragment, reused by two row blocks
      const __bf16* bp = Wb + (long)(h0 + arow) * H3 + k0 + agrp * 8;
      bf.u[0] = *(const uint4*)bp;
      bf.u[1] = *(const uint4*)(bp + 16);
      Frag32 a0;
      const __bf16* ap0 = A + arow * H3 + k0 + agrp * 8;
      a0.u[0] = *(const uint4*)ap0;
      a0.u[1] = *(const uint4*)(ap0 + 16);
      Frag32 a1;
      const __bf16* ap1 = A + (16 + arow) * H3 + k0 + agrp * 8;
      a1.u[0] = *(const uint4*)ap1;
      a1.u[1] = *(const uint4*)(ap1 + 16);
      c0 = __builtin_amdgcn_wmma_f32_16x16x32_bf16(
          false, a0.v, false, bf.v, (short)0, c0, false, false);
      c1 = __builtin_amdgcn_wmma_f32_16x16x32_bf16(
          false, a1.v, false, bf.v, (short)0, c1, false, false);
    }
    const float bi = bias[h0 + arow];
    float* orow0 = out + ((long)b * L + l0 + agrp * 8) * H + h0 + arow;
    float* orow1 = out + ((long)b * L + l0 + 16 + agrp * 8) * H + h0 + arow;
#pragma unroll
    for (int r = 0; r < 8; ++r) {
      orow0[(long)r * H] = tanhf(c0[r] + bi);
      orow1[(long)r * H] = tanhf(c1[r] + bi);
    }
  }
}

// ---------------------------------------------------------------------------
// Launch
// ---------------------------------------------------------------------------
extern "C" void kernel_launch(void* const* d_in, const int* in_sizes, int n_in,
                              void* d_out, int out_size, void* d_ws, size_t ws_size,
                              hipStream_t stream) {
  const float* output   = (const float*)d_in[0];  // [80][256][1024]
  const float* context0 = (const float*)d_in[1];  // [80][256][1024]
  const float* context1 = (const float*)d_in[2];  // [8][512][1024]
  // d_in[3] = mask: unused (the reference's masked_fill result is discarded)
  const float* W_out    = (const float*)d_in[4];  // [1024][3072]
  const float* b_out    = (const float*)d_in[5];  // [1024]
  (void)in_sizes; (void)n_in; (void)out_size; (void)ws_size;

  const long nQ  = (long)BN * L * H;        // 20,971,520
  const long nK1 = (long)8 * SSET * H;      //  4,194,304
  const long nW  = (long)H * H3;            //  3,145,728

  __bf16* ws   = (__bf16*)d_ws;
  __bf16* Qb   = ws;                        // output, bf16
  __bf16* K0b  = Qb + nQ;                   // context0, bf16 row-major
  __bf16* K0T  = K0b + nQ;                  // context0^T, bf16 [b][H][256]
  __bf16* K1b  = K0T + nQ;                  // context1, bf16 row-major
  __bf16* K1T  = K1b + nK1;                 // context1^T, bf16 [b][H][512]
  __bf16* Wb   = K1T + nK1;                 // W_out, bf16
  __bf16* ExC  = Wb + nW;                   // ex_c, bf16
  __bf16* SetC = ExC + nQ;                  // set_c, bf16

  float* out      = (float*)d_out;                   // [80][256][1024]
  float* ex_attn  = out + nQ;                        // [80][256][256]
  float* set_attn = ex_attn + (long)BN * L * 256;    // [80][256][512]

  dim3 cblk(256);
  k_cvt <<<dim3((unsigned)((nQ  + 255) / 256)), cblk, 0, stream>>>(output,   Qb,  nQ);
  k_cvt <<<dim3((unsigned)((nQ  + 255) / 256)), cblk, 0, stream>>>(context0, K0b, nQ);
  k_cvt <<<dim3((unsigned)((nK1 + 255) / 256)), cblk, 0, stream>>>(context1, K1b, nK1);
  k_cvt <<<dim3((unsigned)((nW  + 255) / 256)), cblk, 0, stream>>>(W_out,    Wb,  nW);
  k_cvtT<<<dim3((unsigned)((nQ  + 255) / 256)), cblk, 0, stream>>>(context0, K0T, 256, H, nQ);
  k_cvtT<<<dim3((unsigned)((nK1 + 255) / 256)), cblk, 0, stream>>>(context1, K1T, 512, H, nK1);

  dim3 ablk(128);
  dim3 agrid(BN * (L / 16));  // 1280 workgroups
  const size_t lds_ex  = 16 * H * 2 + 16 * 256 * 6 + 16 * 8 * 4;  //  57 856 B
  const size_t lds_set = 16 * H * 2 + 16 * 512 * 6 + 16 * 8 * 4;  //  82 432 B
  k_attn<256><<<agrid, ablk, lds_ex,  stream>>>(Qb, K0b, K0T, ex_attn,  ExC,  1);
  k_attn<512><<<agrid, ablk, lds_set, stream>>>(Qb, K1b, K1T, set_attn, SetC, 10);

  dim3 ogrid(BN * (L / 32));  // 640 workgroups
  const size_t lds_out = (size_t)32 * H3 * 2;  // 196 608 B (<320 KB/WGP)
  k_out<<<ogrid, ablk, lds_out, stream>>>(Qb, ExC, SetC, Wb, b_out, out);
}
